// Extralonger_52956946760485
// MI455X (gfx1250) — compile-verified, matrix operand order checked by reading, not run on hardware
//
#include <hip/hip_runtime.h>
#include <stdint.h>

// ---------------- constants (from the reference) ----------------
constexpr int Bc = 8, Tc = 12, Nc = 1024, Dc = 512, Hc = 8, FFc = 1024, SPDc = 288;
constexpr int HDc = Dc / Hc;     // 64
constexpr int BT  = Bc * Tc;     // 96
constexpr int BN  = Bc * Nc;     // 8192

typedef __attribute__((ext_vector_type(16))) _Float16 v16h;
typedef __attribute__((ext_vector_type(8)))  float    v8f;

union FragU { uint32_t u[8]; v16h v; };

__device__ __forceinline__ uint32_t pack_f16(float a, float b) {
  union { _Float16 h[2]; uint32_t u; } x;
  x.h[0] = (_Float16)a; x.h[1] = (_Float16)b;
  return x.u;
}

// low 32 bits of a generic pointer to __shared__ == LDS byte offset (ISA: LDS aperture
// mapping discards upper bits)
__device__ __forceinline__ uint32_t lds_off(const void* p) {
  return (uint32_t)(uintptr_t)p;
}

// gfx1250 async global->LDS DMA (CDNA5 ISA); tracked by ASYNCcnt
__device__ __forceinline__ void async_ld_b32(uint32_t lds, const void* g) {
  asm volatile("global_load_async_to_lds_b32 %0, %1, off" :: "v"(lds), "v"(g) : "memory");
}
__device__ __forceinline__ void async_ld_b128(uint32_t lds, const void* g) {
  asm volatile("global_load_async_to_lds_b128 %0, %1, off" :: "v"(lds), "v"(g) : "memory");
}
__device__ __forceinline__ void wait_async0() {
  asm volatile("s_wait_asynccnt 0x0" ::: "memory");
}

// ---------------- WMMA GEMM: C = act(scale*(A@B) + bias) ----------------
// A element (m,k) = A[b*sAb + h*sAh + m*sAm + k]            (k-stride 1 always)
// B element (k,n) = Bw[b*sBb + h*sBh + k*sBk + n*sBn]
// C element (m,n) = C[b*sCb + h*sCh + m*sCm + n]
// Block = 128 threads (4 waves). Block tile = 32(M) x 64(N): each wave owns one 16-col
// strip and two 16x16 accumulators (B fragment reused across 2 WMMAs per k-step).
// Tiles staged f32 in LDS via async-to-LDS DMA; f32->f16 at fragment build.
struct WGemmP {
  const float* A; const float* Bw; const float* bias; float* C;
  int M, N, K, HH;
  long long sAm, sAb, sAh;
  long long sBk, sBn, sBb, sBh;
  long long sCm, sCb, sCh;
  float scale; int act;
};

__global__ __launch_bounds__(128) void wmma_gemm_f16(WGemmP p) {
  __shared__ __align__(16) float AsF[32 * 32];   // 32 rows x 32 k (f32)
  __shared__ __align__(16) float BsF[32 * 64];   // 32 k x 64 cols (k-major, f32)
  const int tid  = threadIdx.x;
  const int wave = tid >> 5, lane = tid & 31;
  const int z = blockIdx.z;
  const int b = z / p.HH, h = z % p.HH;
  const int M0 = blockIdx.y << 5;                // 32 rows per block
  const int N0 = blockIdx.x << 6;                // 64 cols per block
  const float* A  = p.A  + (long long)b * p.sAb + (long long)h * p.sAh;
  const float* Bw = p.Bw + (long long)b * p.sBb + (long long)h * p.sBh;
  const uint32_t aBase = lds_off(AsF);
  const uint32_t bBase = lds_off(BsF);

  v8f acc0 = {0.f,0.f,0.f,0.f,0.f,0.f,0.f,0.f};
  v8f acc1 = {0.f,0.f,0.f,0.f,0.f,0.f,0.f,0.f};

  const int m    = lane & 15;
  const int kb   = (lane >> 4) << 3;
  const int ncol = (wave << 4) + m;

  for (int k0 = 0; k0 < p.K; k0 += 32) {
    // A tile: 32 rows x 32 k = 256 x b128 transfers, 2 per thread (k-contiguous)
    {
      int i = tid << 1;
      #pragma unroll
      for (int t2 = 0; t2 < 2; ++t2, ++i) {
        int row = i >> 3;
        int k4  = (i & 7) << 2;
        const float* g = A + (long long)(M0 + row) * p.sAm + (k0 + k4);
        async_ld_b128(aBase + (uint32_t)(((row << 5) + k4) << 2), g);
      }
    }
    // B tile: 32 k x 64 cols = 2048 x b32 transfers, 16 per thread (stride-generic)
    {
      int i = tid << 4;
      #pragma unroll
      for (int t2 = 0; t2 < 16; ++t2, ++i) {
        int kk  = i >> 6;
        int col = i & 63;
        const float* g = Bw + (long long)(k0 + kk) * p.sBk + (long long)(N0 + col) * p.sBn;
        async_ld_b32(bBase + (uint32_t)(((kk << 6) + col) << 2), g);
      }
    }
    wait_async0();
    __syncthreads();

    // fragments per ISA 16-bit 16x32 layout:
    // lane m (0..15): k 0..7 / 16..23 ; lane m+16: k 8..15 / 24..31
    FragU af0, af1, bf;
    #pragma unroll
    for (int j = 0; j < 8; ++j) {
      int off = kb + ((j & 3) << 1) + ((j >> 2) << 4);   // even k within 0..31
      af0.u[j] = pack_f16(AsF[(m << 5) + off],        AsF[(m << 5) + off + 1]);
      af1.u[j] = pack_f16(AsF[((16 + m) << 5) + off], AsF[((16 + m) << 5) + off + 1]);
      bf.u[j]  = pack_f16(BsF[(off << 6) + ncol],     BsF[((off + 1) << 6) + ncol]);
    }
    acc0 = __builtin_amdgcn_wmma_f32_16x16x32_f16(false, af0.v, false, bf.v, (short)0, acc0, false, false);
    acc1 = __builtin_amdgcn_wmma_f32_16x16x32_f16(false, af1.v, false, bf.v, (short)0, acc1, false, false);
    __syncthreads();
  }

  // store per C layout: vgpr r -> row r (lanes 0-15) / row r+8 (lanes 16-31)
  float* C = p.C + (long long)b * p.sCb + (long long)h * p.sCh;
  const int colBase = N0 + ncol;
  const int rowAdd  = (lane >> 4) << 3;
  const float bv  = p.bias ? p.bias[colBase] : 0.f;
  const float scl = p.scale;
  const bool  act = p.act != 0;
  #pragma unroll
  for (int r = 0; r < 8; ++r) {
    float v0 = acc0[r] * scl + bv;
    float v1 = acc1[r] * scl + bv;
    if (act) { v0 = v0 > 0.f ? v0 : 0.f; v1 = v1 > 0.f ? v1 : 0.f; }
    int row0 = M0 + r + rowAdd;
    C[(long long)row0 * p.sCm + colBase]        = v0;
    C[(long long)(row0 + 16) * p.sCm + colBase] = v1;
  }
}

// ---------------- strided scalar GEMM (small d=12 ops) ----------------
struct SGemmP {
  const float* A; const float* Bw; const float* bias; float* C;
  int M, N, K, HH;
  long long sAm, sAk, sAb, sAh;
  long long sBk, sBn, sBb, sBh;
  long long sCm, sCn, sCb, sCh;
  float scale; int act;
};

__global__ __launch_bounds__(256) void sgemm(SGemmP p) {
  long long idx = (long long)blockIdx.x * blockDim.x + threadIdx.x;
  long long tot = (long long)p.M * p.N;
  if (idx >= tot) return;
  int m = (int)(idx / p.N), n = (int)(idx % p.N);
  int z = blockIdx.z, b = z / p.HH, h = z % p.HH;
  const float* A  = p.A  + (long long)b*p.sAb + (long long)h*p.sAh + (long long)m*p.sAm;
  const float* Bw = p.Bw + (long long)b*p.sBb + (long long)h*p.sBh + (long long)n*p.sBn;
  float bv = p.bias ? p.bias[n] : 0.f;
  float acc = 0.f;
  for (int k = 0; k < p.K; ++k) acc += A[(long long)k*p.sAk] * Bw[(long long)k*p.sBk];
  acc = acc * p.scale + bv;
  if (p.act) acc = acc > 0.f ? acc : 0.f;
  p.C[(long long)b*p.sCb + (long long)h*p.sCh + (long long)m*p.sCm + (long long)n*p.sCn] = acc;
}

// ---------------- block reductions (blockDim.x == 256) ----------------
__device__ __forceinline__ float bred_sum(float v, float* red) {
  int t = threadIdx.x;
  red[t] = v; __syncthreads();
  for (int s = 128; s > 0; s >>= 1) { if (t < s) red[t] += red[t+s]; __syncthreads(); }
  float r = red[0]; __syncthreads(); return r;
}
__device__ __forceinline__ float bred_max(float v, float* red) {
  int t = threadIdx.x;
  red[t] = v; __syncthreads();
  for (int s = 128; s > 0; s >>= 1) { if (t < s) red[t] = fmaxf(red[t], red[t+s]); __syncthreads(); }
  float r = red[0]; __syncthreads(); return r;
}

// ---------------- residual + LayerNorm: x = LN(x + res)*g + b ----------------
__global__ __launch_bounds__(256) void ln_res(float* x, const float* res,
                                              const float* g, const float* bt, int W) {
  __shared__ float red[256];
  long long row = blockIdx.x;
  float* xr = x + row * W; const float* rr = res + row * W;
  int t = threadIdx.x;
  float s = 0.f, s2 = 0.f;
  for (int i = t; i < W; i += 256) { float v = xr[i] + rr[i]; s += v; s2 += v*v; }
  s  = bred_sum(s,  red);
  s2 = bred_sum(s2, red);
  float mean = s / W;
  float var  = s2 / W - mean*mean;
  float rinv = rsqrtf(var + 1e-5f);
  for (int i = t; i < W; i += 256) {
    float v = xr[i] + rr[i];
    xr[i] = (v - mean) * rinv * g[i] + bt[i];
  }
}

// ---------------- softmax over rows ----------------
__global__ __launch_bounds__(256) void softmax_row(float* sc, int W) {
  __shared__ float red[256];
  float* r = sc + (long long)blockIdx.x * W;
  int t = threadIdx.x;
  float mx = -1e30f;
  for (int i = t; i < W; i += 256) mx = fmaxf(mx, r[i]);
  mx = bred_max(mx, red);
  float sum = 0.f;
  for (int i = t; i < W; i += 256) sum += expf(r[i] - mx);
  sum = bred_sum(sum, red);
  float inv = 1.f / sum;
  for (int i = t; i < W; i += 256) r[i] = expf(r[i] - mx) * inv;
}

// dual softmax for layer_s: r = 0.5*(softmax(r) + softmax(r masked by adj[l]))
__global__ __launch_bounds__(256) void softmax_dual(float* sc, const unsigned char* adj, int L) {
  __shared__ float red[256];
  int row = blockIdx.x;
  int l   = row % L;
  float* r = sc + (long long)row * L;
  const unsigned char* am = adj + (long long)l * L;
  int t = threadIdx.x;
  float gmx = -1e30f, mmx = -1e30f;
  for (int i = t; i < L; i += 256) {
    float v = r[i];
    gmx = fmaxf(gmx, v);
    if (am[i]) mmx = fmaxf(mmx, v);
  }
  gmx = bred_max(gmx, red);
  mmx = bred_max(mmx, red);
  float gs = 0.f, ms = 0.f;
  for (int i = t; i < L; i += 256) {
    float v = r[i];
    gs += expf(v - gmx);
    if (am[i]) ms += expf(v - mmx);
  }
  gs = bred_sum(gs, red);
  ms = bred_sum(ms, red);
  float gi = 1.f / gs, mi = 1.f / ms;
  for (int i = t; i < L; i += 256) {
    float v = r[i];
    float a  = expf(v - gmx) * gi;
    float lo = am[i] ? expf(v - mmx) * mi : 0.f;
    r[i] = 0.5f * (a + lo);
  }
}

// ---------------- fused tiny attention (L=12, hd=64) ----------------
__global__ __launch_bounds__(64) void attn_small(const float* q, const float* k, const float* v,
                                                 float* out, int L, int H, int hd, int D, float scale) {
  __shared__ float s[16];
  int bid = blockIdx.x;
  int l = bid % L; int h = (bid / L) % H; int b = bid / (L * H);
  int j = threadIdx.x;
  const float* qrow = q + ((long long)b * L + l) * D + h * hd;
  if (j < L) {
    const float* krow = k + ((long long)b * L + j) * D + h * hd;
    float acc = 0.f;
    for (int c = 0; c < hd; ++c) acc += qrow[c] * krow[c];
    s[j] = acc * scale;
  }
  __syncthreads();
  float mx = -1e30f;
  for (int m = 0; m < L; ++m) mx = fmaxf(mx, s[m]);
  float sum = 0.f, o = 0.f;
  for (int m = 0; m < L; ++m) {
    float e = expf(s[m] - mx);
    sum += e;
    o += e * v[((long long)b * L + m) * D + h * hd + j];
  }
  out[((long long)b * L + l) * D + h * hd + j] = o / sum;
}

// ---------------- elementwise / embedding kernels ----------------
__global__ void embed_xd_k(const float* x, const float* noise, float* xd) {
  long long idx = (long long)blockIdx.x * 256 + threadIdx.x;
  if (idx >= (long long)BT * Nc) return;
  int n = (int)(idx % Nc); int t = (int)((idx / Nc) % Tc);
  xd[idx] = x[idx * 3] + noise[t * Nc + n];
}

__global__ void embed_tod_dow_k(const float* x, const float* todE, const float* dowE, float* x_t) {
  int bt = blockIdx.x; int c = threadIdx.x;   // 96 blocks x 128 threads
  float tod = x[(((long long)bt * Nc) + 0) * 3 + 1];
  float dow = x[(((long long)bt * Nc) + 0) * 3 + 2];
  int ti = (int)(tod * SPDc); if (ti < 0) ti = 0; if (ti >= SPDc) ti = SPDc - 1;
  int di = (int)dow;          if (di < 0) di = 0; if (di > 6) di = 6;
  x_t[(long long)bt * Dc + 256 + c] = todE[ti * 128 + c];
  x_t[(long long)bt * Dc + 384 + c] = dowE[di * 128 + c];
}

__global__ void copy_k(const float* src, float* dst, long long n) {
  long long i = (long long)blockIdx.x * 256 + threadIdx.x;
  if (i < n) dst[i] = src[i];
}

__global__ void s_input_node_k(const float* xd, const float* Ws, const float* bs,
                               const float* nodeE, float* x_s) {
  long long idx = (long long)blockIdx.x * 256 + threadIdx.x;
  if (idx >= (long long)BN * Dc) return;
  int c = (int)(idx % Dc);
  long long bn = idx / Dc;
  int n = (int)(bn % Nc); int b = (int)(bn / Nc);
  float vv;
  if (c < 256) {
    float acc = bs[c];
    for (int t = 0; t < Tc; ++t)
      acc += xd[((long long)b * Tc + t) * Nc + n] * Ws[t * 256 + c];
    vv = acc;
  } else {
    vv = nodeE[(long long)n * 256 + (c - 256)];
  }
  x_s[idx] = vv;
}

__global__ void transpose_mix_k(const float* xm, float* ymat) {
  long long idx = (long long)blockIdx.x * 256 + threadIdx.x;
  if (idx >= (long long)Bc * Dc * Tc) return;
  int t = (int)(idx % Tc); int i = (int)((idx / Tc) % Dc); int b = (int)(idx / ((long long)Tc * Dc));
  ymat[idx] = xm[((long long)b * Tc + t) * Dc + i];
}

__global__ void combine_k(const float* xm, const float* xs, const float* xt, float* out) {
  long long i = (long long)blockIdx.x * 256 + threadIdx.x;
  if (i >= (long long)BT * Nc) return;
  out[i] = (2.f * xm[i] + xs[i] + xt[i]) * 0.25f;
}

// ---------------- host helpers ----------------
struct SalP { const float *qw,*qb,*kw,*kb,*vw,*vb,*ow,*ob,*f1w,*f1b,*f2w,*f2b,*g1,*b1,*g2,*b2; };
static SalP salAt(void* const* in, int base) {
  SalP s;
  const float** f = (const float**)&s;
  for (int i = 0; i < 16; ++i) f[i] = (const float*)in[base + i];
  return s;
}

static void gemmW(hipStream_t st, const float* A, const float* Bw, const float* bias, float* C,
                  int M, int N, int K, int Z, int HH,
                  long long sAm, long long sAb, long long sAh,
                  long long sBk, long long sBn, long long sBb, long long sBh,
                  long long sCm, long long sCb, long long sCh,
                  float scale, int act) {
  WGemmP p{A, Bw, bias, C, M, N, K, HH, sAm, sAb, sAh, sBk, sBn, sBb, sBh, sCm, sCb, sCh, scale, act};
  dim3 g((unsigned)(N / 64), (unsigned)(M / 32), (unsigned)Z);
  wmma_gemm_f16<<<g, 128, 0, st>>>(p);
}

static void gemmS(hipStream_t st, const float* A, const float* Bw, const float* bias, float* C,
                  int M, int N, int K, int Z, int HH,
                  long long sAm, long long sAk, long long sAb, long long sAh,
                  long long sBk, long long sBn, long long sBb, long long sBh,
                  long long sCm, long long sCn, long long sCb, long long sCh,
                  float scale, int act) {
  SGemmP p{A, Bw, bias, C, M, N, K, HH, sAm, sAk, sAb, sAh, sBk, sBn, sBb, sBh, sCm, sCn, sCb, sCh, scale, act};
  dim3 g((unsigned)(((long long)M * N + 255) / 256), 1, (unsigned)Z);
  sgemm<<<g, 256, 0, st>>>(p);
}

// full SAL block for d=512, L=12 sequences (layer_t / layer_mix_t), x: 96x512 in-place
static void sal_small(hipStream_t st, float* xb, const SalP& P,
                      float* sq, float* sk, float* sv, float* sattn, float* sh) {
  gemmW(st, xb, P.qw, P.qb, sq, BT, Dc, Dc, 1, 1, Dc,0,0, Dc,1,0,0, Dc,0,0, 1.f, 0);
  gemmW(st, xb, P.kw, P.kb, sk, BT, Dc, Dc, 1, 1, Dc,0,0, Dc,1,0,0, Dc,0,0, 1.f, 0);
  gemmW(st, xb, P.vw, P.vb, sv, BT, Dc, Dc, 1, 1, Dc,0,0, Dc,1,0,0, Dc,0,0, 1.f, 0);
  attn_small<<<Bc * Hc * Tc, HDc, 0, st>>>(sq, sk, sv, sattn, Tc, Hc, HDc, Dc, 0.125f);
  gemmW(st, sattn, P.ow, P.ob, sq, BT, Dc, Dc, 1, 1, Dc,0,0, Dc,1,0,0, Dc,0,0, 1.f, 0);
  ln_res<<<BT, 256, 0, st>>>(xb, sq, P.g1, P.b1, Dc);
  gemmW(st, xb, P.f1w, P.f1b, sh, BT, FFc, Dc, 1, 1, Dc,0,0, FFc,1,0,0, FFc,0,0, 1.f, 1);
  gemmW(st, sh, P.f2w, P.f2b, sk, BT, Dc, FFc, 1, 1, FFc,0,0, Dc,1,0,0, Dc,0,0, 1.f, 0);
  ln_res<<<BT, 256, 0, st>>>(xb, sk, P.g2, P.b2, Dc);
}

// big SAL (layer_s): x_s 8192x512 in-place, adjacency dual softmax; per-batch score chunk
static void sal_big(hipStream_t st, float* xs, const SalP& P, const unsigned char* adj,
                    float* qb_, float* kb_, float* vb_, float* ab_, float* hb_, float* chunk) {
  const long long LD = (long long)Nc * Dc;
  gemmW(st, xs, P.qw, P.qb, qb_, BN, Dc, Dc, 1, 1, Dc,0,0, Dc,1,0,0, Dc,0,0, 1.f, 0);
  gemmW(st, xs, P.kw, P.kb, kb_, BN, Dc, Dc, 1, 1, Dc,0,0, Dc,1,0,0, Dc,0,0, 1.f, 0);
  gemmW(st, xs, P.vw, P.vb, vb_, BN, Dc, Dc, 1, 1, Dc,0,0, Dc,1,0,0, Dc,0,0, 1.f, 0);
  for (int b = 0; b < Bc; ++b) {
    // scores[h][l][m] = (Q . K)/8   (B read transposed via strides)
    gemmW(st, qb_ + b * LD, kb_ + b * LD, nullptr, chunk,
          Nc, Nc, HDc, Hc, Hc,
          Dc, 0, HDc,
          1, Dc, 0, HDc,
          Nc, 0, (long long)Nc * Nc,
          0.125f, 0);
    softmax_dual<<<Hc * Nc, 256, 0, st>>>(chunk, adj, Nc);
    // attn_out[l][h*64+j] = A @ V
    gemmW(st, chunk, vb_ + b * LD, nullptr, ab_ + b * LD,
          Nc, HDc, Nc, Hc, Hc,
          Nc, 0, (long long)Nc * Nc,
          Dc, 1, 0, HDc,
          Dc, 0, HDc,
          1.f, 0);
  }
  gemmW(st, ab_, P.ow, P.ob, qb_, BN, Dc, Dc, 1, 1, Dc,0,0, Dc,1,0,0, Dc,0,0, 1.f, 0);
  ln_res<<<BN, 256, 0, st>>>(xs, qb_, P.g1, P.b1, Dc);
  gemmW(st, xs, P.f1w, P.f1b, hb_, BN, FFc, Dc, 1, 1, Dc,0,0, FFc,1,0,0, FFc,0,0, 1.f, 1);
  gemmW(st, hb_, P.f2w, P.f2b, kb_, BN, Dc, FFc, 1, 1, FFc,0,0, Dc,1,0,0, Dc,0,0, 1.f, 0);
  ln_res<<<BN, 256, 0, st>>>(xs, kb_, P.g2, P.b2, Dc);
}

// ---------------- entry ----------------
extern "C" void kernel_launch(void* const* d_in, const int* in_sizes, int n_in,
                              void* d_out, int out_size, void* d_ws, size_t ws_size,
                              hipStream_t stream) {
  (void)in_sizes; (void)n_in; (void)out_size; (void)ws_size;
  const float* x     = (const float*)d_in[0];
  const unsigned char* adj = (const unsigned char*)d_in[1];   // jnp bool -> 1 byte
  const float* noise = (const float*)d_in[2];
  const float* todE  = (const float*)d_in[3];
  const float* dowE  = (const float*)d_in[4];
  const float* nodeE = (const float*)d_in[5];
  const float* tiw = (const float*)d_in[6];  const float* tib = (const float*)d_in[7];
  const float* tow = (const float*)d_in[8];  const float* tob = (const float*)d_in[9];
  const float* siw = (const float*)d_in[10]; const float* sib = (const float*)d_in[11];
  const float* sow = (const float*)d_in[12]; const float* sob = (const float*)d_in[13];
  const float* m1w = (const float*)d_in[14]; const float* m1b = (const float*)d_in[15];
  const float* m2w = (const float*)d_in[16]; const float* m2b = (const float*)d_in[17];
  SalP Lt  = salAt(d_in, 18);
  SalP Ls  = salAt(d_in, 34);
  SalP Lmt = salAt(d_in, 50);
  SalP Lms = salAt(d_in, 66);
  float* out = (float*)d_out;

  // workspace carve (~172 MB total)
  char* wp = (char*)d_ws;
  auto alloc = [&](size_t nf) { float* r = (float*)wp; wp += ((nf * sizeof(float) + 255) / 256) * 256; return r; };
  float* xd        = alloc((size_t)BT * Nc);
  float* x_t       = alloc((size_t)BT * Dc);
  float* x_mix     = alloc((size_t)BT * Dc);
  float* x_t_out   = alloc((size_t)BT * Nc);
  float* x_mix_out = alloc((size_t)BT * Nc);
  float* x_s_out   = alloc((size_t)BT * Nc);
  float* x_s   = alloc((size_t)BN * Dc);
  float* qbuf  = alloc((size_t)BN * Dc);
  float* kbuf  = alloc((size_t)BN * Dc);
  float* vbuf  = alloc((size_t)BN * Dc);
  float* abuf  = alloc((size_t)BN * Dc);
  float* hbuf  = alloc((size_t)BN * FFc);
  float* chunk = alloc((size_t)Hc * Nc * Nc);        // per-batch score chunk (also reused by mix_s)
  float* sq = alloc((size_t)BT * Dc); float* sk = alloc((size_t)BT * Dc);
  float* sv = alloc((size_t)BT * Dc); float* sattn = alloc((size_t)BT * Dc);
  float* sh = alloc((size_t)BT * FFc);
  float* ymat  = alloc((size_t)Bc * Dc * Tc);        // (b, i=0..511, t=0..11) contiguous
  float* mq = alloc((size_t)Bc * Dc * Tc); float* mk = alloc((size_t)Bc * Dc * Tc);
  float* mv = alloc((size_t)Bc * Dc * Tc); float* mattn = alloc((size_t)Bc * Dc * Tc);
  float* mo = alloc((size_t)Bc * Dc * Tc);
  float* mh = alloc((size_t)Bc * Dc * FFc);
  float* mf = alloc((size_t)Bc * Dc * Tc);
  float* xm1 = alloc((size_t)BT * Nc);

  // 1. xd = x[...,0] + input_noise
  embed_xd_k<<<(BT * Nc + 255) / 256, 256, 0, stream>>>(x, noise, xd);
  // 2. x_t[:, :256] = xd @ t_input (WMMA, M=96,K=1024,N=256)
  gemmW(stream, xd, tiw, tib, x_t, BT, 256, Nc, 1, 1, Nc,0,0, 256,1,0,0, Dc,0,0, 1.f, 0);
  // 3. x_t[:, 256:] = tod/dow embeddings
  embed_tod_dow_k<<<BT, 128, 0, stream>>>(x, todE, dowE, x_t);
  // 4. x_mix = x_t
  copy_k<<<(BT * Dc + 255) / 256, 256, 0, stream>>>(x_t, x_mix, (long long)BT * Dc);
  // 5. x_s = [xd^T @ s_input , node_emb]
  s_input_node_k<<<(int)(((long long)BN * Dc + 255) / 256), 256, 0, stream>>>(xd, siw, sib, nodeE, x_s);

  // 6. temporal branch: layer_t then t_output (WMMA M=96,K=512,N=1024)
  sal_small(stream, x_t, Lt, sq, sk, sv, sattn, sh);
  gemmW(stream, x_t, tow, tob, x_t_out, BT, Nc, Dc, 1, 1, Dc,0,0, Nc,1,0,0, Nc,0,0, 1.f, 0);

  // 7. mix branch: layer_mix_t (d=512) then layer_mix_s (d=12, seq=512, heads=2)
  sal_small(stream, x_mix, Lmt, sq, sk, sv, sattn, sh);
  transpose_mix_k<<<(int)(((long long)Bc * Dc * Tc + 255) / 256), 256, 0, stream>>>(x_mix, ymat);
  gemmS(stream, ymat, Lms.qw, Lms.qb, mq, Bc*Dc, Tc, Tc, 1, 1, Tc,1,0,0, Tc,1,0,0, Tc,1,0,0, 1.f, 0);
  gemmS(stream, ymat, Lms.kw, Lms.kb, mk, Bc*Dc, Tc, Tc, 1, 1, Tc,1,0,0, Tc,1,0,0, Tc,1,0,0, 1.f, 0);
  gemmS(stream, ymat, Lms.vw, Lms.vb, mv, Bc*Dc, Tc, Tc, 1, 1, Tc,1,0,0, Tc,1,0,0, Tc,1,0,0, 1.f, 0);
  // scores (z = b*2+h), hd=6
  gemmS(stream, mq, mk, nullptr, chunk, Dc, Dc, 6, Bc*2, 2,
        Tc,1,(long long)Dc*Tc,6,  1,Tc,(long long)Dc*Tc,6,
        Dc,1,(long long)2*Dc*Dc,(long long)Dc*Dc, 0.40824829046386f, 0);
  softmax_row<<<Bc * 2 * Dc, 256, 0, stream>>>(chunk, Dc);
  gemmS(stream, chunk, mv, nullptr, mattn, Dc, 6, Dc, Bc*2, 2,
        Dc,1,(long long)2*Dc*Dc,(long long)Dc*Dc,  Tc,1,(long long)Dc*Tc,6,
        Tc,1,(long long)Dc*Tc,6, 1.f, 0);
  gemmS(stream, mattn, Lms.ow, Lms.ob, mo, Bc*Dc, Tc, Tc, 1, 1, Tc,1,0,0, Tc,1,0,0, Tc,1,0,0, 1.f, 0);
  ln_res<<<Bc * Dc, 256, 0, stream>>>(ymat, mo, Lms.g1, Lms.b1, Tc);
  gemmS(stream, ymat, Lms.f1w, Lms.f1b, mh, Bc*Dc, FFc, Tc, 1, 1, Tc,1,0,0, FFc,1,0,0, FFc,1,0,0, 1.f, 1);
  gemmS(stream, mh, Lms.f2w, Lms.f2b, mf, Bc*Dc, Tc, FFc, 1, 1, FFc,1,0,0, Tc,1,0,0, Tc,1,0,0, 1.f, 0);
  ln_res<<<Bc * Dc, 256, 0, stream>>>(ymat, mf, Lms.g2, Lms.b2, Tc);
  // mix_proj1: (B,T,512)@ (512->1024)   (A = ymat viewed transposed via strides)
  gemmS(stream, ymat, m1w, m1b, xm1, Tc, Nc, Dc, Bc, 1,
        1,Tc,(long long)Dc*Tc,0,  Nc,1,0,0,
        Nc,1,(long long)Tc*Nc,0, 1.f, 0);
  // mix_proj2 + final swap -> x_mix_out[b,t2,n]
  gemmS(stream, xm1, m2w, m2b, x_mix_out, Nc, Tc, Tc, Bc, 1,
        1,Nc,(long long)Tc*Nc,0,  Tc,1,0,0,
        1,Nc,(long long)Tc*Nc,0, 1.f, 0);

  // 8. spatial branch: layer_s (big WMMA SAL with dual softmax), then s_output (+swap)
  sal_big(stream, x_s, Ls, adj, qbuf, kbuf, vbuf, abuf, hbuf, chunk);
  gemmS(stream, x_s, sow, sob, x_s_out, Nc, Tc, Dc, Bc, 1,
        Dc,1,(long long)Nc*Dc,0,  Tc,1,0,0,
        1,Nc,(long long)Tc*Nc,0, 1.f, 0);

  // 9. out = (2*x_mix + x_s + x_t)/4
  combine_k<<<(BT * Nc + 255) / 256, 256, 0, stream>>>(x_mix_out, x_s_out, x_t_out, out);
}
